// EfficientAttention_16243566313525
// MI455X (gfx1250) — matrix operand order repaired and checked by reading, not compiled, not run on hardware
//
#include <hip/hip_runtime.h>

// ---------------------------------------------------------------------------
// Problem constants (from reference): N=8, C=C2=256, H=W=96 -> L=9216
// heads=8, key_dim=value_dim=32
// ---------------------------------------------------------------------------
static constexpr int    Lq   = 9216;     // spatial positions (96*96)
static constexpr int    Cch  = 256;      // channels
static constexpr int    Nbat = 8;        // batch
static constexpr size_t Sel  = (size_t)Nbat * Cch * Lq;   // 18,874,368 elems

// WMMA vector types (probe-confirmed signatures)
typedef __attribute__((ext_vector_type(16))) __bf16 v16bf;
typedef __attribute__((ext_vector_type(8)))  float  v8f;

union BFrag {
    v16bf v;
    unsigned int u[8];
    unsigned short s[16];
    uint4 q[2];
};

__device__ inline unsigned short f2bf(float x) {
    unsigned int u = __float_as_uint(x);
    unsigned int r = u + 0x7FFFu + ((u >> 16) & 1u);   // round-to-nearest-even
    return (unsigned short)(r >> 16);
}
__device__ inline float bf2f(unsigned short s) {
    return __uint_as_float(((unsigned int)s) << 16);
}

// ---------------------------------------------------------------------------
// 1) BatchNorm statistics: one block per (tensor, channel).
//    scales[t*256+c] = gamma*rstd ; scales[512+t*256+c] = beta - mean*scale
// ---------------------------------------------------------------------------
__global__ void __launch_bounds__(256)
bn_stats_kernel(const float* __restrict__ Q, const float* __restrict__ KV,
                const float* __restrict__ qg, const float* __restrict__ qb,
                const float* __restrict__ kg, const float* __restrict__ kb,
                float* __restrict__ scales)
{
    const int blk = blockIdx.x;
    const int t = blk >> 8;            // 0 = q_input, 1 = kv
    const int c = blk & 255;
    const float* X = t ? KV : Q;
    const int tid = threadIdx.x;

    float s = 0.f, ss = 0.f;
    for (int n = 0; n < Nbat; ++n) {
        const float* p = X + ((size_t)n * Cch + c) * Lq;
        for (int l = tid * 4; l < Lq; l += 1024) {
            float4 v4 = *(const float4*)(p + l);
            s  += v4.x + v4.y + v4.z + v4.w;
            ss += v4.x * v4.x + v4.y * v4.y + v4.z * v4.z + v4.w * v4.w;
        }
    }
    __shared__ float rs[256], rss[256];
    rs[tid] = s; rss[tid] = ss;
    __syncthreads();
    for (int o = 128; o > 0; o >>= 1) {
        if (tid < o) { rs[tid] += rs[tid + o]; rss[tid] += rss[tid + o]; }
        __syncthreads();
    }
    if (tid == 0) {
        const float inv_n = 1.0f / (float)(Nbat * Lq);
        float mean = rs[0] * inv_n;
        float var  = rss[0] * inv_n - mean * mean;
        float g  = t ? kg[c] : qg[c];
        float be = t ? kb[c] : qb[c];
        float sc = g * rsqrtf(var + 0.001f);
        scales[t * 256 + c]       = sc;
        scales[512 + t * 256 + c] = be - mean * sc;
    }
}

// ---------------------------------------------------------------------------
// 2) Convert the four 256x256 f32 weight matrices to bf16 (contiguous output)
// ---------------------------------------------------------------------------
__global__ void __launch_bounds__(256)
cvt_weights_kernel(const float* __restrict__ a, const float* __restrict__ b,
                   const float* __restrict__ c, const float* __restrict__ d,
                   unsigned short* __restrict__ o)
{
    int i = blockIdx.x * 256 + threadIdx.x;       // 4*65536 total
    int w = i >> 16, j = i & 65535;
    const float* s = (w == 0) ? a : (w == 1) ? b : (w == 2) ? c : d;
    o[i] = f2bf(s[j]);
}

// ---------------------------------------------------------------------------
// 3) Apply BN affine + convert to bf16; 8 elements per thread (vectorized).
//    blockIdx.y selects tensor.  L (9216) is a multiple of 8, so an 8-chunk
//    never straddles a channel row.
// ---------------------------------------------------------------------------
__global__ void __launch_bounds__(256)
normalize_kernel(const float* __restrict__ Q, const float* __restrict__ KV,
                 const float* __restrict__ scales,
                 unsigned short* __restrict__ Qn, unsigned short* __restrict__ KVn)
{
    const int t = blockIdx.y;
    size_t i8 = ((size_t)blockIdx.x * 256 + threadIdx.x) * 8;    // < Sel
    int c = (int)((i8 / Lq) & 255);
    float sc = scales[t * 256 + c];
    float bi = scales[512 + t * 256 + c];
    const float* src = (t ? KV : Q) + i8;
    float4 v0 = *(const float4*)(src);
    float4 v1 = *(const float4*)(src + 4);
    union { uint4 q4; unsigned short s[8]; } o;
    o.s[0] = f2bf(v0.x * sc + bi); o.s[1] = f2bf(v0.y * sc + bi);
    o.s[2] = f2bf(v0.z * sc + bi); o.s[3] = f2bf(v0.w * sc + bi);
    o.s[4] = f2bf(v1.x * sc + bi); o.s[5] = f2bf(v1.y * sc + bi);
    o.s[6] = f2bf(v1.z * sc + bi); o.s[7] = f2bf(v1.w * sc + bi);
    *(uint4*)((t ? KVn : Qn) + i8) = o.q4;
}

// ---------------------------------------------------------------------------
// 4) Tiled WMMA GEMM:  Out[n][m][l] = sum_k W[m][k] * X[n][k][l]
//    Block tile 64(M) x 256(N), K-step 32; 8 waves -> wave tile 32 x 64.
//    LDS K-stride padded to 40 ushorts (80 B): multiple of 16 B so all
//    fragment accesses are ds_*_b128, and 20 mod 64 spreads 16 lanes over
//    16 distinct banks (conflict-free).
//    RESID=false: bf16 store.  RESID=true: f32 store + residual add.
// ---------------------------------------------------------------------------
static constexpr int BM = 64, BN = 256, BK = 32, BKP = 40;

template <bool RESID>
__global__ void __launch_bounds__(256)
gemm_wmma_kernel(const unsigned short* __restrict__ W,
                 const unsigned short* __restrict__ X,
                 unsigned short* __restrict__ Ob,
                 float* __restrict__ Of,
                 const float* __restrict__ R)
{
    __shared__ unsigned short Alds[BM][BKP];   // [m][k]
    __shared__ unsigned short Bt[BN][BKP];     // [l][k]  (transposed stage)

    const int tid  = threadIdx.x;
    const int lane = tid & 31;
    const int wave = tid >> 5;
    const int wm   = (wave >> 2) * 32;          // 0 or 32
    const int wn   = (wave & 3) * 64;           // 0,64,128,192
    const int lblk = blockIdx.x * BN;
    const int mblk = blockIdx.y * BM;
    const int n    = blockIdx.z;
    const unsigned short* Xn = X + (size_t)n * Cch * Lq;
    const int half = lane >> 4;

    v8f acc[2][4];
    for (int i = 0; i < 2; ++i)
        for (int j = 0; j < 4; ++j)
            for (int e = 0; e < 8; ++e) acc[i][j][e] = 0.f;

    for (int k0 = 0; k0 < Cch; k0 += BK) {
        // stage A tile (64x32): one b128 global load + one b128 LDS store
        {
            int m  = tid >> 2;
            int kq = (tid & 3) << 3;
            *(uint4*)&Alds[m][kq] =
                *(const uint4*)&W[(size_t)(mblk + m) * Cch + k0 + kq];
        }
        // stage B tile (32x256) transposed into Bt[l][k]: read two k-rows as
        // uint4 (coalesced along l), store packed k-pairs as aligned b32.
        for (int q = 0; q < 2; ++q) {
            int unit = tid + q * 256;            // 0..511 (16 k-pairs x 32 l-chunks)
            int kp = unit >> 5;                  // 0..15
            int lc = (unit & 31) << 3;           // 0..248 step 8
            int k  = kp << 1;
            union { uint4 q4; unsigned short s[8]; } r0, r1;
            r0.q4 = *(const uint4*)&Xn[(size_t)(k0 + k)     * Lq + lblk + lc];
            r1.q4 = *(const uint4*)&Xn[(size_t)(k0 + k + 1) * Lq + lblk + lc];
            for (int i = 0; i < 8; ++i) {
                unsigned int pk = (unsigned int)r0.s[i] |
                                  ((unsigned int)r1.s[i] << 16);
                *(unsigned int*)&Bt[lc + i][k] = pk;
            }
        }
        __syncthreads();

        // A fragments (16-bit A 16x32 layout, ISA 7.12.2) -> 2x ds_load_b128
        BFrag a[2];
        for (int mi = 0; mi < 2; ++mi) {
            int row = wm + mi * 16 + (lane & 15);
            a[mi].q[0] = *(const uint4*)&Alds[row][half << 3];
            a[mi].q[1] = *(const uint4*)&Alds[row][16 + (half << 3)];
        }
        // B fragments + WMMA
        for (int ni = 0; ni < 4; ++ni) {
            BFrag b;
            int col  = wn + ni * 16 + (lane & 15);
            int koff = half << 3;
            b.q[0] = *(const uint4*)&Bt[col][koff];
            b.q[1] = *(const uint4*)&Bt[col][koff + 16];
            for (int mi = 0; mi < 2; ++mi)
                acc[mi][ni] = __builtin_amdgcn_wmma_f32_16x16x32_bf16(
                    false, a[mi].v, false, b.v, (short)0, acc[mi][ni], false, false);
        }
        __syncthreads();
    }

    // store (C/D 16x16 f32 layout: VGPR r -> M = r + 8*half, lane&15 -> N)
    for (int mi = 0; mi < 2; ++mi) {
        for (int ni = 0; ni < 4; ++ni) {
            int colg = lblk + wn + ni * 16 + (lane & 15);
            for (int r = 0; r < 8; ++r) {
                int rowg = mblk + wm + mi * 16 + r + (half << 3);
                size_t idx = ((size_t)n * Cch + rowg) * Lq + colg;
                float v = acc[mi][ni][r];
                if (RESID) Of[idx] = v + R[idx];
                else       Ob[idx] = f2bf(v);
            }
        }
    }
}

// ---------------------------------------------------------------------------
// 5) In-place softmax over L for each key row (2048 rows of 9216 bf16)
// ---------------------------------------------------------------------------
__global__ void __launch_bounds__(256)
softmax_rows_kernel(unsigned short* __restrict__ Kp)
{
    unsigned short* p = Kp + (size_t)blockIdx.x * Lq;
    const int tid = threadIdx.x;
    __shared__ float red[256];

    float m = -1e30f;
    for (int l = tid; l < Lq; l += 256) m = fmaxf(m, bf2f(p[l]));
    red[tid] = m; __syncthreads();
    for (int o = 128; o > 0; o >>= 1) {
        if (tid < o) red[tid] = fmaxf(red[tid], red[tid + o]);
        __syncthreads();
    }
    float mx = red[0];
    __syncthreads();

    float s = 0.f;
    for (int l = tid; l < Lq; l += 256) s += __expf(bf2f(p[l]) - mx);
    red[tid] = s; __syncthreads();
    for (int o = 128; o > 0; o >>= 1) {
        if (tid < o) red[tid] += red[tid + o];
        __syncthreads();
    }
    float inv = 1.f / red[0];

    for (int l = tid; l < Lq; l += 256)
        p[l] = f2bf(__expf(bf2f(p[l]) - mx) * inv);
}

// ---------------------------------------------------------------------------
// 6) Query softmax over the 32 key_dim channels at fixed (n,h,l). In place.
// ---------------------------------------------------------------------------
__global__ void __launch_bounds__(256)
qsoftmax_kernel(unsigned short* __restrict__ Qp)
{
    size_t t = (size_t)blockIdx.x * 256 + threadIdx.x;    // 64*L total
    if (t >= (size_t)64 * Lq) return;
    int l  = (int)(t % Lq);
    int nh = (int)(t / Lq);
    unsigned short* p = Qp + ((size_t)(nh >> 3) * Cch + (nh & 7) * 32) * Lq + l;

    float m = -1e30f;
    for (int k = 0; k < 32; ++k) m = fmaxf(m, bf2f(p[(size_t)k * Lq]));
    float s = 0.f;
    for (int k = 0; k < 32; ++k) s += __expf(bf2f(p[(size_t)k * Lq]) - m);
    float inv = 1.f / s;
    for (int k = 0; k < 32; ++k)
        p[(size_t)k * Lq] = f2bf(__expf(bf2f(p[(size_t)k * Lq]) - m) * inv);
}

// ---------------------------------------------------------------------------
// 7) context[n][h][k][v] = sum_l key_sm[k][l] * values[v][l]
//    One block per (n,h); 8 waves split the L range, 2x2 frags each,
//    cross-wave reduce in LDS.  Stored TRANSPOSED: ctxT[n][h][v][k].
// ---------------------------------------------------------------------------
__global__ void __launch_bounds__(256)
context_wmma_kernel(const unsigned short* __restrict__ Ksm,
                    const unsigned short* __restrict__ V,
                    float* __restrict__ ctxT)
{
    __shared__ float part[8][32][32];
    const int tid = threadIdx.x, lane = tid & 31, wave = tid >> 5;
    const int nh = blockIdx.x;                    // 0..63
    const size_t base = ((size_t)(nh >> 3) * Cch + (nh & 7) * 32) * Lq;
    const int half = lane >> 4;

    v8f acc[2][2];
    for (int i = 0; i < 2; ++i)
        for (int j = 0; j < 2; ++j)
            for (int e = 0; e < 8; ++e) acc[i][j][e] = 0.f;

    const int span = Lq / 8;                      // 1152 per wave
    const int lstart = wave * span;
    for (int l0 = lstart; l0 < lstart + span; l0 += 32) {
        BFrag a[2], b[2];
        for (int mi = 0; mi < 2; ++mi) {          // m = key_dim row
            int row = mi * 16 + (lane & 15);
            const unsigned short* kp = Ksm + base + (size_t)row * Lq + l0;
            for (int j = 0; j < 8; ++j) {
                int kb = (half << 4) + (j << 1);
                a[mi].u[j] = *(const unsigned int*)(kp + kb);
            }
        }
        for (int ni = 0; ni < 2; ++ni) {          // n = value_dim column
            int col = ni * 16 + (lane & 15);
            const unsigned short* vp = V + base + (size_t)col * Lq + l0 + (half << 4);
            for (int j = 0; j < 8; ++j)
                b[ni].u[j] = *(const unsigned int*)(vp + (j << 1));
        }
        for (int mi = 0; mi < 2; ++mi)
            for (int ni = 0; ni < 2; ++ni)
                acc[mi][ni] = __builtin_amdgcn_wmma_f32_16x16x32_bf16(
                    false, a[mi].v, false, b[ni].v, (short)0, acc[mi][ni], false, false);
    }

    for (int mi = 0; mi < 2; ++mi)
        for (int ni = 0; ni < 2; ++ni)
            for (int r = 0; r < 8; ++r)
                part[wave][mi * 16 + r + (half << 3)][ni * 16 + (lane & 15)] = acc[mi][ni][r];
    __syncthreads();

    for (int t = tid; t < 1024; t += 256) {
        int kk = t >> 5, vv = t & 31;
        float s = 0.f;
        for (int w = 0; w < 8; ++w) s += part[w][kk][vv];
        ctxT[((size_t)nh * 32 + vv) * 32 + kk] = s;   // transposed: [v][k]
    }
}

// ---------------------------------------------------------------------------
// 8) attended[n][h][v][l] = sum_k ctxT[v][k] * query_sm[k][l]
//    K = 32 -> single WMMA step per 16x16 tile.  bf16 output.
// ---------------------------------------------------------------------------
__global__ void __launch_bounds__(256)
attended_wmma_kernel(const float* __restrict__ ctxT,
                     const unsigned short* __restrict__ Qsm,
                     unsigned short* __restrict__ Att)
{
    const int tid = threadIdx.x, lane = tid & 31, wave = tid >> 5;
    const int nh = blockIdx.y;
    const int lblk = blockIdx.x * 512 + wave * 64;
    const int half = lane >> 4;
    const float* cp = ctxT + (size_t)nh * 1024;
    const size_t qbase = ((size_t)(nh >> 3) * Cch + (nh & 7) * 32) * Lq;

    // A = ctxT (M=v, K=k), converted f32->bf16 on the fly
    BFrag a[2];
    for (int mi = 0; mi < 2; ++mi) {
        int row = mi * 16 + (lane & 15);
        for (int j = 0; j < 8; ++j) {
            int kb = (half << 4) + (j << 1);
            unsigned int lo = f2bf(cp[row * 32 + kb]);
            unsigned int hi = f2bf(cp[row * 32 + kb + 1]);
            a[mi].u[j] = lo | (hi << 16);
        }
    }

    for (int ni = 0; ni < 4; ++ni) {
        int col = lblk + ni * 16 + (lane & 15);
        BFrag b;
        int koff = half << 4;
        for (int j = 0; j < 16; ++j)
            b.s[j] = Qsm[qbase + (size_t)(koff + j) * Lq + col];
        v8f zero;
        for (int e = 0; e < 8; ++e) zero[e] = 0.f;
        for (int mi = 0; mi < 2; ++mi) {
            v8f d = __builtin_amdgcn_wmma_f32_16x16x32_bf16(
                false, a[mi].v, false, b.v, (short)0, zero, false, false);
            for (int r = 0; r < 8; ++r) {
                int vg = mi * 16 + r + (half << 3);
                Att[qbase + (size_t)vg * Lq + col] = f2bf(d[r]);
            }
        }
    }
}

// ---------------------------------------------------------------------------
// Launch
// ---------------------------------------------------------------------------
extern "C" void kernel_launch(void* const* d_in, const int* in_sizes, int n_in,
                              void* d_out, int out_size, void* d_ws, size_t ws_size,
                              hipStream_t stream)
{
    (void)in_sizes; (void)n_in; (void)out_size; (void)ws_size;
    const float* q_input = (const float*)d_in[0];
    const float* kv      = (const float*)d_in[1];
    const float* Wq      = (const float*)d_in[2];
    const float* Wk      = (const float*)d_in[3];
    const float* Wv      = (const float*)d_in[4];
    const float* Wp      = (const float*)d_in[5];
    const float* qg      = (const float*)d_in[6];
    const float* qb      = (const float*)d_in[7];
    const float* kg      = (const float*)d_in[8];
    const float* kb      = (const float*)d_in[9];

    char* ws = (char*)d_ws;
    // workspace layout (bytes)
    constexpr size_t OFF_SCALES = 0;                        // 1024 f32
    constexpr size_t OFF_CTX    = 4096;                     // 65536 f32
    constexpr size_t OFF_WB     = OFF_CTX + 262144;         // 4 x 65536 bf16
    constexpr size_t OFF_QN     = OFF_WB + 4 * 65536 * 2;
    constexpr size_t OFF_KVN    = OFF_QN + Sel * 2;
    constexpr size_t OFF_QP     = OFF_KVN + Sel * 2;
    constexpr size_t OFF_KP     = OFF_QP + Sel * 2;
    constexpr size_t OFF_VP     = OFF_KP + Sel * 2;

    float*          scales = (float*)(ws + OFF_SCALES);
    float*          ctxT   = (float*)(ws + OFF_CTX);
    unsigned short* Wqb    = (unsigned short*)(ws + OFF_WB);
    unsigned short* Wkb    = Wqb + 65536;
    unsigned short* Wvb    = Wqb + 2 * 65536;
    unsigned short* Wpb    = Wqb + 3 * 65536;
    unsigned short* Qn     = (unsigned short*)(ws + OFF_QN);
    unsigned short* KVn    = (unsigned short*)(ws + OFF_KVN);
    unsigned short* Qp     = (unsigned short*)(ws + OFF_QP);
    unsigned short* Kp     = (unsigned short*)(ws + OFF_KP);
    unsigned short* Vp     = (unsigned short*)(ws + OFF_VP);
    unsigned short* Att    = KVn;   // kv-normalized buffer is dead by then

    // 1) BN stats (512 blocks: 2 tensors x 256 channels)
    bn_stats_kernel<<<512, 256, 0, stream>>>(q_input, kv, qg, qb, kg, kb, scales);
    // 2) weights -> bf16
    cvt_weights_kernel<<<1024, 256, 0, stream>>>(Wq, Wk, Wv, Wp, Wqb);
    // 3) normalize + bf16 (8 elems/thread)
    normalize_kernel<<<dim3((unsigned)(Sel / 8 / 256), 2), 256, 0, stream>>>(
        q_input, kv, scales, Qn, KVn);
    // 4) projections (L/BN=36, C/BM=4, batches=8)
    dim3 ggrid(Lq / BN, Cch / BM, Nbat);
    gemm_wmma_kernel<false><<<ggrid, 256, 0, stream>>>(Wqb, Qn,  Qp, nullptr, nullptr);
    gemm_wmma_kernel<false><<<ggrid, 256, 0, stream>>>(Wkb, KVn, Kp, nullptr, nullptr);
    gemm_wmma_kernel<false><<<ggrid, 256, 0, stream>>>(Wvb, KVn, Vp, nullptr, nullptr);
    // 5) key softmax over L (2048 rows), in place
    softmax_rows_kernel<<<2048, 256, 0, stream>>>(Kp);
    // 6) query softmax over 32 channels, in place
    qsoftmax_kernel<<<(unsigned)((64ull * Lq) / 256), 256, 0, stream>>>(Qp);
    // 7) per-head context (64 blocks), stored transposed
    context_wmma_kernel<<<64, 256, 0, stream>>>(Kp, Vp, ctxT);
    // 8) attended (18 l-chunks x 64 head-batches)
    attended_wmma_kernel<<<dim3(Lq / 512, 64), 256, 0, stream>>>(ctxT, Qp, Att);
    // 9) reprojection + residual -> f32 output
    gemm_wmma_kernel<true><<<ggrid, 256, 0, stream>>>(Wpb, Att, nullptr,
                                                      (float*)d_out, q_input);
}
